// TransformerEncoder_7267084665178
// MI455X (gfx1250) — matrix operand order repaired and checked by reading, not compiled, not run on hardware
//
#include <hip/hip_runtime.h>
#include <hip/hip_bf16.h>

#ifndef __has_builtin
#define __has_builtin(x) 0
#endif

typedef __attribute__((ext_vector_type(16))) _Float16 v16h;
typedef __attribute__((ext_vector_type(8)))  _Float16 v8h;
typedef __attribute__((ext_vector_type(8)))  float    v8f;
typedef __attribute__((ext_vector_type(4)))  int      v4i;

#define AS1 __attribute__((address_space(1)))
#define AS3 __attribute__((address_space(3)))

#if __has_builtin(__builtin_amdgcn_global_load_async_to_lds_b128) && \
    __has_builtin(__builtin_amdgcn_s_wait_asynccnt)
#define USE_ASYNC_LDS 1
#endif

// ---- problem constants (match reference) ----
#define BQ    2
#define LQ    13294
#define DQ    256
#define HQ    8
#define NLQ   4
#define NPQ   4
#define CQ    32
#define DFFQ  1024
#define NLAY  6
#define MTOT  (BQ * LQ)          // 26588 rows total

#define BM 128
#define BN 64
#define BK 32
#define LPAD 40                  // LDS row stride in halves (80 B, 16B-aligned)

__device__ __forceinline__ v16h cat8(v8h lo, v8h hi) {
  return __builtin_shufflevector(lo, hi,
      0, 1, 2, 3, 4, 5, 6, 7, 8, 9, 10, 11, 12, 13, 14, 15);
}

// ---------------------------------------------------------------------------
// f32 [K][N] -> f16 transposed [N][K], for all layers of one weight type.
// ---------------------------------------------------------------------------
__global__ void cvt_transpose_kernel(const float* __restrict__ s,
                                     _Float16* __restrict__ d,
                                     int K, int N, int nlay) {
  size_t per = (size_t)K * N;
  size_t i = (size_t)blockIdx.x * blockDim.x + threadIdx.x;
  if (i >= per * (size_t)nlay) return;
  size_t l = i / per, r = i % per;
  int k = (int)(r / N), n = (int)(r % N);
  d[l * per + (size_t)n * K + k] = (_Float16)s[i];
}

// q = f16(cur + pos), x = f16(cur)
__global__ void add_convert_kernel(const float* __restrict__ cur,
                                   const float* __restrict__ pos,
                                   _Float16* __restrict__ qh,
                                   _Float16* __restrict__ xh, size_t n) {
  size_t i = (size_t)blockIdx.x * blockDim.x + threadIdx.x;
  if (i < n) {
    float c = cur[i];
    qh[i] = (_Float16)(c + pos[i]);
    xh[i] = (_Float16)c;
  }
}

// ---------------------------------------------------------------------------
// WMMA GEMM: out[M,N] = A_f16[M,K] @ W_f16T[N,K]^T + bias
//   mode 0: store f32 | mode 1: relu->f16 | mode 2: f32 with row mask zero
// Block tile 128x64, 8 waves = 4(M) x 2(N), each wave 32x32 via 4 WMMAs.
// Double-buffered LDS, async global->LDS DMA pipelined one tile ahead.
// ---------------------------------------------------------------------------
__global__ __launch_bounds__(256)
void gemm_f16_kernel(const _Float16* __restrict__ A,
                     const _Float16* __restrict__ WtT,
                     const float* __restrict__ bias,
                     float* __restrict__ outF,
                     _Float16* __restrict__ outH,
                     const unsigned char* __restrict__ rowmask,
                     int Mdim, int Ndim, int Kdim, int mode) {
  __shared__ _Float16 lA[2][BM][LPAD];   // 2 x 128 x 32 used
  __shared__ _Float16 lB[2][BN][LPAD];   // 2 x  64 x 32 used

  const int tid  = threadIdx.x;
  const int lane = tid & 31;
  const int wave = tid >> 5;            // 8 waves
  const int wy   = wave & 3;            // M sub-tile (4 x 32 rows)
  const int wx   = wave >> 2;           // N sub-tile (2 x 32 cols)
  const int r    = lane & 15;
  const int sh   = lane >> 4;

  const int m0 = blockIdx.x * BM;
  const int n0 = blockIdx.y * BN;

  // staging: A tile = 4096 halves (2 chunks/thread), B tile = 2048 (1 chunk)
  const int srow  = (tid * 8) >> 5;     // 0..63
  const int scol  = (tid * 8) & 31;
  const int rowA0 = srow;
  const int rowA1 = srow + 64;
  const bool vA0 = (m0 + rowA0) < Mdim;
  const bool vA1 = (m0 + rowA1) < Mdim;

  v8f acc00 = {}, acc01 = {}, acc10 = {}, acc11 = {};

  const int nT = Kdim / BK;

  auto stageTile = [&](int t) {
    const int k0 = t * BK;
    const int bs = t & 1;
    const _Float16* sa0 = A   + (size_t)(m0 + rowA0) * Kdim + k0 + scol;
    const _Float16* sa1 = A   + (size_t)(m0 + rowA1) * Kdim + k0 + scol;
    const _Float16* sb  = WtT + (size_t)(n0 + srow)  * Kdim + k0 + scol;
    if (vA0) {
#ifdef USE_ASYNC_LDS
      __builtin_amdgcn_global_load_async_to_lds_b128(
          (AS1 v4i*)sa0, (AS3 v4i*)&lA[bs][rowA0][scol], 0, 0);
#else
      *(v8h*)&lA[bs][rowA0][scol] = *(const v8h*)sa0;
#endif
    } else {
      v8h z = {};
      *(v8h*)&lA[bs][rowA0][scol] = z;
    }
    if (vA1) {
#ifdef USE_ASYNC_LDS
      __builtin_amdgcn_global_load_async_to_lds_b128(
          (AS1 v4i*)sa1, (AS3 v4i*)&lA[bs][rowA1][scol], 0, 0);
#else
      *(v8h*)&lA[bs][rowA1][scol] = *(const v8h*)sa1;
#endif
    } else {
      v8h z = {};
      *(v8h*)&lA[bs][rowA1][scol] = z;
    }
#ifdef USE_ASYNC_LDS
    __builtin_amdgcn_global_load_async_to_lds_b128(
        (AS1 v4i*)sb, (AS3 v4i*)&lB[bs][srow][scol], 0, 0);
#else
    *(v8h*)&lB[bs][srow][scol] = *(const v8h*)sb;
#endif
    if (k0 + 2 * BK < Kdim) {           // global_prefetch_b8 two stages ahead
      if (vA0) __builtin_prefetch(sa0 + 2 * BK, 0, 3);
      __builtin_prefetch(sb + 2 * BK, 0, 3);
    }
  };

  // prologue: stage tile 0
  stageTile(0);
#ifdef USE_ASYNC_LDS
  __builtin_amdgcn_s_wait_asynccnt(0);
#endif
  __syncthreads();

  for (int t = 0; t < nT; ++t) {
    const int cb = t & 1;
    if (t + 1 < nT) stageTile(t + 1);   // DMA next tile into alternate buffer

    // --- fragments (ISA 7.12.2): all b128 LDS loads, 16B aligned ---
    v16h a0 = cat8(*(const v8h*)&lA[cb][wy * 32 + r][sh * 8],
                   *(const v8h*)&lA[cb][wy * 32 + r][16 + sh * 8]);
    v16h a1 = cat8(*(const v8h*)&lA[cb][wy * 32 + 16 + r][sh * 8],
                   *(const v8h*)&lA[cb][wy * 32 + 16 + r][16 + sh * 8]);
    const v8h* pb0 = (const v8h*)&lB[cb][wx * 32 + r][0];
    v16h b0 = cat8(pb0[sh], pb0[2 + sh]);
    const v8h* pb1 = (const v8h*)&lB[cb][wx * 32 + 16 + r][0];
    v16h b1 = cat8(pb1[sh], pb1[2 + sh]);

    acc00 = __builtin_amdgcn_wmma_f32_16x16x32_f16(false, a0, false, b0,
                                                   (short)0, acc00, false, false);
    acc01 = __builtin_amdgcn_wmma_f32_16x16x32_f16(false, a0, false, b1,
                                                   (short)0, acc01, false, false);
    acc10 = __builtin_amdgcn_wmma_f32_16x16x32_f16(false, a1, false, b0,
                                                   (short)0, acc10, false, false);
    acc11 = __builtin_amdgcn_wmma_f32_16x16x32_f16(false, a1, false, b1,
                                                   (short)0, acc11, false, false);

#ifdef USE_ASYNC_LDS
    __builtin_amdgcn_s_wait_asynccnt(0);
#endif
    __syncthreads();
  }

  // --- epilogue; C/D layout: element v -> M = sh*8+v, N = r ---
  auto store = [&](const v8f& acc, int msub, int nsub) {
#pragma unroll
    for (int v = 0; v < 8; ++v) {
      int gm = m0 + wy * 32 + msub * 16 + sh * 8 + v;
      if (gm >= Mdim) continue;
      float mk = 1.0f;
      if (mode == 2 && rowmask != nullptr && rowmask[gm]) mk = 0.0f;
      int gn = n0 + wx * 32 + nsub * 16 + r;
      float c = (acc[v] + bias[gn]) * mk;
      if (mode == 1) outH[(size_t)gm * Ndim + gn] = (_Float16)fmaxf(c, 0.0f);
      else           outF[(size_t)gm * Ndim + gn] = c;
    }
  };
  store(acc00, 0, 0);
  store(acc01, 0, 1);
  store(acc10, 1, 0);
  store(acc11, 1, 1);
}

// ---------------------------------------------------------------------------
// softmax over the 16 (NL*NP) attention logits per (m, head); in place
// ---------------------------------------------------------------------------
__global__ void softmax16_kernel(float* __restrict__ att, int total) {
  int t = blockIdx.x * blockDim.x + threadIdx.x;
  if (t >= total) return;
  int m = t / HQ, h = t % HQ;
  float* a = att + (size_t)m * (HQ * 16) + h * 16;
  float mx = a[0];
#pragma unroll
  for (int j = 1; j < 16; ++j) mx = fmaxf(mx, a[j]);
  float e[16], sum = 0.0f;
#pragma unroll
  for (int j = 0; j < 16; ++j) { e[j] = __expf(a[j] - mx); sum += e[j]; }
  float inv = 1.0f / sum;
#pragma unroll
  for (int j = 0; j < 16; ++j) a[j] = e[j] * inv;
}

// ---------------------------------------------------------------------------
// Multi-scale deformable sampling. One thread per (m, head).
// ---------------------------------------------------------------------------
__global__ void sampling_kernel(const float* __restrict__ value,
                                const float* __restrict__ off,
                                const float* __restrict__ att,
                                const float* __restrict__ vr,
                                _Float16* __restrict__ sampH) {
  int t = blockIdx.x * blockDim.x + threadIdx.x;
  if (t >= MTOT * HQ) return;
  int m = t / HQ, h = t % HQ;
  int b = m / LQ, l = m % LQ;

  int lq, idx;
  if (l < 10000)      { lq = 0; idx = l; }
  else if (l < 12500) { lq = 1; idx = l - 10000; }
  else if (l < 13125) { lq = 2; idx = l - 12500; }
  else                { lq = 3; idx = l - 13125; }
  const int lw[4] = {100, 50, 25, 13};
  const int lh[4] = {100, 50, 25, 13};
  const int ls[4] = {0, 10000, 12500, 13125};
  int wq = lw[lq], hq = lh[lq];
  int row = idx / wq, col = idx % wq;
  float vrxq = vr[(b * NLQ + lq) * 2 + 0];
  float vryq = vr[(b * NLQ + lq) * 2 + 1];
  float rx = ((float)col + 0.5f) / (vrxq * (float)wq);
  float ry = ((float)row + 0.5f) / (vryq * (float)hq);

  const float* offp = off + (size_t)m * 256 + h * 32;
  const float* attp = att + (size_t)m * 128 + h * 16;

  float acc[CQ];
#pragma unroll
  for (int c = 0; c < CQ; ++c) acc[c] = 0.0f;

#pragma unroll
  for (int lvl = 0; lvl < NLQ; ++lvl) {
    int w = lw[lvl], hh = lh[lvl];
    float vrx = vr[(b * NLQ + lvl) * 2 + 0];
    float vry = vr[(b * NLQ + lvl) * 2 + 1];
    float refx = rx * vrx, refy = ry * vry;
    const float* vbase = value + ((size_t)(b * LQ + ls[lvl])) * DQ + h * CQ;
#pragma unroll
    for (int p = 0; p < NPQ; ++p) {
      float ox = offp[(lvl * NPQ + p) * 2 + 0];
      float oy = offp[(lvl * NPQ + p) * 2 + 1];
      float aw = attp[lvl * NPQ + p];
      float gx = (refx + ox / (float)w) * (float)w - 0.5f;
      float gy = (refy + oy / (float)hh) * (float)hh - 0.5f;
      float x0f = floorf(gx), y0f = floorf(gy);
      int x0 = (int)x0f, y0 = (int)y0f;
      float wx1 = gx - x0f, wy1 = gy - y0f;
      float w00 = (1.0f - wx1) * (1.0f - wy1);
      float w10 = wx1 * (1.0f - wy1);
      float w01 = (1.0f - wx1) * wy1;
      float w11 = wx1 * wy1;
      bool vx0 = (x0 >= 0) && (x0 < w);
      bool vx1 = (x0 + 1 >= 0) && (x0 + 1 < w);
      bool vy0 = (y0 >= 0) && (y0 < hh);
      bool vy1 = (y0 + 1 >= 0) && (y0 + 1 < hh);
      int xc0 = min(max(x0, 0), w - 1),  xc1 = min(max(x0 + 1, 0), w - 1);
      int yc0 = min(max(y0, 0), hh - 1), yc1 = min(max(y0 + 1, 0), hh - 1);
      float f00 = (vx0 && vy0) ? aw * w00 : 0.0f;
      float f10 = (vx1 && vy0) ? aw * w10 : 0.0f;
      float f01 = (vx0 && vy1) ? aw * w01 : 0.0f;
      float f11 = (vx1 && vy1) ? aw * w11 : 0.0f;
      const float* p00 = vbase + (size_t)(yc0 * w + xc0) * DQ;
      const float* p10 = vbase + (size_t)(yc0 * w + xc1) * DQ;
      const float* p01 = vbase + (size_t)(yc1 * w + xc0) * DQ;
      const float* p11 = vbase + (size_t)(yc1 * w + xc1) * DQ;
#pragma unroll
      for (int c = 0; c < CQ; ++c)
        acc[c] += f00 * p00[c] + f10 * p10[c] + f01 * p01[c] + f11 * p11[c];
    }
  }
  _Float16* o = sampH + (size_t)m * DQ + h * CQ;
#pragma unroll
  for (int c = 0; c < CQ; ++c) o[c] = (_Float16)acc[c];
}

// ---------------------------------------------------------------------------
// out = LN(cur + delta) * g + beta ; optional f16 copy for next GEMM input
// ---------------------------------------------------------------------------
__global__ __launch_bounds__(256)
void ln_kernel(float* __restrict__ cur, const float* __restrict__ delta,
               const float* __restrict__ g, const float* __restrict__ bta,
               _Float16* __restrict__ outH, int writeH) {
  __shared__ float red[256];
  int rowi = blockIdx.x, tid = threadIdx.x;
  size_t i = (size_t)rowi * DQ + tid;
  float x = cur[i] + delta[i];
  red[tid] = x;
  __syncthreads();
#pragma unroll
  for (int o = 128; o > 0; o >>= 1) {
    if (tid < o) red[tid] += red[tid + o];
    __syncthreads();
  }
  float mean = red[0] * (1.0f / 256.0f);
  __syncthreads();
  float d = x - mean;
  red[tid] = d * d;
  __syncthreads();
#pragma unroll
  for (int o = 128; o > 0; o >>= 1) {
    if (tid < o) red[tid] += red[tid + o];
    __syncthreads();
  }
  float var = red[0] * (1.0f / 256.0f);
  float y = d * rsqrtf(var + 1e-5f) * g[tid] + bta[tid];
  cur[i] = y;
  if (writeH) outH[i] = (_Float16)y;
}

// ---------------------------------------------------------------------------
extern "C" void kernel_launch(void* const* d_in, const int* in_sizes, int n_in,
                              void* d_out, int out_size, void* d_ws, size_t ws_size,
                              hipStream_t stream) {
  (void)in_sizes; (void)n_in; (void)out_size; (void)ws_size;

  const float* src   = (const float*)d_in[0];
  const float* pos   = (const float*)d_in[1];
  const float* vr    = (const float*)d_in[2];
  const unsigned char* mask = (const unsigned char*)d_in[3];
  const float* Wv    = (const float*)d_in[4];
  const float* bv    = (const float*)d_in[5];
  const float* Woff  = (const float*)d_in[6];
  const float* boff  = (const float*)d_in[7];
  const float* Watt  = (const float*)d_in[8];
  const float* batt  = (const float*)d_in[9];
  const float* Wout  = (const float*)d_in[10];
  const float* bout  = (const float*)d_in[11];
  const float* g1    = (const float*)d_in[12];
  const float* beta1 = (const float*)d_in[13];
  const float* W1    = (const float*)d_in[14];
  const float* b1    = (const float*)d_in[15];
  const float* W2    = (const float*)d_in[16];
  const float* b2    = (const float*)d_in[17];
  const float* g2    = (const float*)d_in[18];
  const float* beta2 = (const float*)d_in[19];

  const size_t nMD = (size_t)MTOT * DQ;
  const size_t nMF = (size_t)MTOT * DFFQ;

  const size_t OFS_WV   = 0;
  const size_t OFS_WOFF = OFS_WV   + (size_t)NLAY * DQ * DQ;
  const size_t OFS_WATT = OFS_WOFF + (size_t)NLAY * DQ * DQ;
  const size_t OFS_WOUT = OFS_WATT + (size_t)NLAY * DQ * 128;
  const size_t OFS_W1   = OFS_WOUT + (size_t)NLAY * DQ * DQ;
  const size_t OFS_W2   = OFS_W1   + (size_t)NLAY * DQ * DFFQ;
  const size_t WT_HALVES= OFS_W2   + (size_t)NLAY * DFFQ * DQ;

  char* ws = (char*)d_ws;
  size_t o = 0;
  _Float16* wtH  = (_Float16*)(ws + o); o += WT_HALVES * 2;
  _Float16* qH   = (_Float16*)(ws + o); o += nMD * 2;
  _Float16* xH   = (_Float16*)(ws + o); o += nMD * 2;
  float*    offB = (float*)   (ws + o); o += nMD * 4;
  float*    attB = (float*)   (ws + o); o += (size_t)MTOT * 128 * 4;
  float*    valB = (float*)   (ws + o); o += nMD * 4;
  _Float16* spH  = (_Float16*)(ws + o); o += nMD * 2;
  float*    tmpB = (float*)   (ws + o); o += nMD * 4;
  _Float16* hidH = (_Float16*)(ws + o); o += nMF * 2;

  float* cur = (float*)d_out;
  (void)hipMemcpyAsync(cur, src, nMD * sizeof(float),
                       hipMemcpyDeviceToDevice, stream);

  auto cvtT = [&](const float* s, _Float16* d, int K, int N) {
    size_t n = (size_t)NLAY * K * N;
    cvt_transpose_kernel<<<(unsigned)((n + 511) / 512), 512, 0, stream>>>(
        s, d, K, N, NLAY);
  };
  cvtT(Wv,   wtH + OFS_WV,   DQ,   DQ);
  cvtT(Woff, wtH + OFS_WOFF, DQ,   DQ);
  cvtT(Watt, wtH + OFS_WATT, DQ,   128);
  cvtT(Wout, wtH + OFS_WOUT, DQ,   DQ);
  cvtT(W1,   wtH + OFS_W1,   DQ,   DFFQ);
  cvtT(W2,   wtH + OFS_W2,   DFFQ, DQ);

  auto gemm = [&](const _Float16* A, const _Float16* Wm, const float* bias,
                  float* oF, _Float16* oH, const unsigned char* rm,
                  int N, int K, int mode) {
    dim3 g((MTOT + BM - 1) / BM, N / BN);
    gemm_f16_kernel<<<g, 256, 0, stream>>>(A, Wm, bias, oF, oH, rm,
                                           MTOT, N, K, mode);
  };

  const int cvtBlocks = (int)((nMD + 511) / 512);
  const int smBlocks  = (MTOT * HQ + 255) / 256;
  const int spBlocks  = (MTOT * HQ + 127) / 128;

  for (int li = 0; li < NLAY; ++li) {
    const _Float16* wv_l   = wtH + OFS_WV   + (size_t)li * DQ * DQ;
    const _Float16* woff_l = wtH + OFS_WOFF + (size_t)li * DQ * DQ;
    const _Float16* watt_l = wtH + OFS_WATT + (size_t)li * DQ * 128;
    const _Float16* wout_l = wtH + OFS_WOUT + (size_t)li * DQ * DQ;
    const _Float16* w1_l   = wtH + OFS_W1   + (size_t)li * DQ * DFFQ;
    const _Float16* w2_l   = wtH + OFS_W2   + (size_t)li * DFFQ * DQ;
    const float* bv_l   = bv   + (size_t)li * DQ;
    const float* boff_l = boff + (size_t)li * DQ;
    const float* batt_l = batt + (size_t)li * 128;
    const float* bout_l = bout + (size_t)li * DQ;
    const float* b1_l   = b1   + (size_t)li * DFFQ;
    const float* b2_l   = b2   + (size_t)li * DQ;
    const float* g1_l   = g1   + (size_t)li * DQ;
    const float* be1_l  = beta1+ (size_t)li * DQ;
    const float* g2_l   = g2   + (size_t)li * DQ;
    const float* be2_l  = beta2+ (size_t)li * DQ;

    add_convert_kernel<<<cvtBlocks, 512, 0, stream>>>(cur, pos, qH, xH, nMD);

    gemm(xH, wv_l, bv_l, valB, nullptr, mask, DQ, DQ, 2);
    gemm(qH, woff_l, boff_l, offB, nullptr, nullptr, DQ, DQ, 0);
    gemm(qH, watt_l, batt_l, attB, nullptr, nullptr, 128, DQ, 0);
    softmax16_kernel<<<smBlocks, 256, 0, stream>>>(attB, MTOT * HQ);

    sampling_kernel<<<spBlocks, 128, 0, stream>>>(valB, offB, attB, vr, spH);

    gemm(spH, wout_l, bout_l, tmpB, nullptr, nullptr, DQ, DQ, 0);
    ln_kernel<<<MTOT, 256, 0, stream>>>(cur, tmpB, g1_l, be1_l, xH, 1);

    gemm(xH, w1_l, b1_l, nullptr, hidH, nullptr, DFFQ, DQ, 1);
    gemm(hidH, w2_l, b2_l, tmpB, nullptr, nullptr, DQ, DFFQ, 0);
    ln_kernel<<<MTOT, 256, 0, stream>>>(cur, tmpB, g2_l, be2_l, nullptr, 0);
  }
}
